// DoubleAttention_34995393528472
// MI455X (gfx1250) — matrix operand order repaired and checked by reading, not compiled, not run on hardware
//
#include <hip/hip_runtime.h>

// ---------------------------------------------------------------------------
// Double Attention (A^2-Net) fused pipeline for MI455X (gfx1250, wave32).
// All GEMMs use v_wmma_f32_16x16x32_f16 (f16 in, f32 accumulate).
// Memory-bound problem: x read once, intermediates kept in f16, no atomics.
// ---------------------------------------------------------------------------

typedef __attribute__((ext_vector_type(16))) _Float16 v16h;
typedef __attribute__((ext_vector_type(8)))  float    v8f;

union Frag16 { v16h h; unsigned u[8]; };

#define C_IN  512
#define NPIX  4096
#define CM    128
#define NP1   32        // pixels per workgroup in projection pass
#define XS_STR 520      // padded half-stride for x tile (bank-conflict avoidance)
#define BM_STR 132      // padded f32 stride for softmax staging
#define NP2   64        // pixels per workgroup in output pass
#define VM_STR 136      // padded half-stride for Vm/Z tiles

__device__ __forceinline__ v8f wmma_f16(v16h a, v16h b, v8f c) {
  return __builtin_amdgcn_wmma_f32_16x16x32_f16(false, a, false, b, (short)0, c,
                                                false, false);
}

// A-operand (16x32, 16-bit): lanes 0-15 hold K=0..15, lanes 16-31 hold K=16..31
__device__ __forceinline__ v16h load_fragA(const unsigned* u32, size_t halfBase, int g) {
  Frag16 f;
#pragma unroll
  for (int v = 0; v < 8; ++v) {
    int kk = 2 * v + 16 * g;
    f.u[v] = u32[(halfBase + (size_t)kk) >> 1];
  }
  return f.h;
}

// B-operand (32x16, 16-bit): lanes 0-15 hold K=0..15, lanes 16-31 hold K=16..31
__device__ __forceinline__ v16h load_fragB(const unsigned* u32, size_t halfBase, int g) {
  Frag16 f;
#pragma unroll
  for (int v = 0; v < 8; ++v) {
    int kk = 2 * v + 16 * g;
    f.u[v] = u32[(halfBase + (size_t)kk) >> 1];
  }
  return f.h;
}

// ---------------------------------------------------------------------------
// Pass 0: convert weights f32 -> f16 into workspace (row-major, same layout)
// ---------------------------------------------------------------------------
__global__ __launch_bounds__(256) void kprep(
    const float* __restrict__ WA, const float* __restrict__ WB,
    const float* __restrict__ WV, const float* __restrict__ WR,
    _Float16* __restrict__ WAh, _Float16* __restrict__ WBh,
    _Float16* __restrict__ WVh, _Float16* __restrict__ WRh)
{
  int i = blockIdx.x * 256 + threadIdx.x;
  if (i < CM * C_IN) {
    WAh[i] = (_Float16)WA[i];
    WBh[i] = (_Float16)WB[i];
    WVh[i] = (_Float16)WV[i];
    WRh[i] = (_Float16)WR[i];
  }
}

// ---------------------------------------------------------------------------
// Pass 1a: projections A, Bm=softmax_c(WB x), Vm=softmax_c(WV x); f16 outputs.
// One workgroup = one (batch, 32-pixel tile). 8 waves, wave = 16-row m-stripe.
// ---------------------------------------------------------------------------
__device__ __forceinline__ void proj2(const unsigned* Wu, const unsigned* xu,
                                      int m0, int lm, int g, v8f& a0, v8f& a1) {
  v8f z = {0.f,0.f,0.f,0.f,0.f,0.f,0.f,0.f};
  a0 = z; a1 = z;
  for (int k0 = 0; k0 < C_IN; k0 += 32) {
    v16h wf = load_fragA(Wu, (size_t)(m0 + lm) * C_IN + (size_t)k0, g);
    v16h x0 = load_fragB(xu, (size_t)lm * XS_STR + (size_t)k0, g);
    v16h x1 = load_fragB(xu, (size_t)(16 + lm) * XS_STR + (size_t)k0, g);
    a0 = wmma_f16(wf, x0, a0);
    a1 = wmma_f16(wf, x1, a1);
  }
}

__device__ __forceinline__ void softmax32x128_store(
    float* bmS, float* red, int tid,
    _Float16* __restrict__ dst, int bb, int n0)
{
  const int p = tid & 31;      // pixel
  const int q = tid >> 5;      // 8 groups of 16 channels each
  float* row = bmS + p * BM_STR + q * 16;
  float mx = -3.402823466e38f;
#pragma unroll
  for (int i = 0; i < 16; ++i) mx = fmaxf(mx, row[i]);
  red[q * 32 + p] = mx;
  __syncthreads();
  if (q == 0) {
    float m2 = red[p];
    for (int j = 1; j < 8; ++j) m2 = fmaxf(m2, red[j * 32 + p]);
    red[p] = m2;
  }
  __syncthreads();
  const float rm = red[p];
  float s = 0.f;
#pragma unroll
  for (int i = 0; i < 16; ++i) { float e = __expf(row[i] - rm); row[i] = e; s += e; }
  red[256 + q * 32 + p] = s;
  __syncthreads();
  if (q == 0) {
    float t = 0.f;
    for (int j = 0; j < 8; ++j) t += red[256 + j * 32 + p];
    red[256 + p] = t;
  }
  __syncthreads();
  const float inv = 1.f / red[256 + p];
#pragma unroll
  for (int i = 0; i < 16; ++i) row[i] *= inv;
  __syncthreads();
  for (int e = tid; e < NP1 * CM; e += 256) {
    int m = e >> 5, j = e & 31;
    dst[((size_t)bb * CM + m) * NPIX + n0 + j] = (_Float16)bmS[j * BM_STR + m];
  }
}

__global__ __launch_bounds__(256) void k1a_proj(
    const float* __restrict__ x,
    const float* __restrict__ bA, const float* __restrict__ bB,
    const float* __restrict__ bV,
    const _Float16* __restrict__ WAh, const _Float16* __restrict__ WBh,
    const _Float16* __restrict__ WVh,
    _Float16* __restrict__ Ag, _Float16* __restrict__ Bmg,
    _Float16* __restrict__ Vmg)
{
  __shared__ __align__(16) _Float16 xs[NP1 * XS_STR];   // 33,280 B
  __shared__ __align__(16) float    bmS[NP1 * BM_STR];  // 16,896 B
  __shared__ __align__(16) float    red[512];           //  2,048 B

  const int blk = blockIdx.x;
  const int bb  = blk >> 7;                // 128 tiles per batch
  const int n0  = (blk & 127) * NP1;
  const int tid = threadIdx.x;
  const int lane = tid & 31;
  const int w    = tid >> 5;
  const int lm   = lane & 15;
  const int g    = lane >> 4;
  const int m0   = w * 16;

  // Stage x tile f32->f16, pixel-major (coalesced reads along n)
  for (int e = tid; e < NP1 * C_IN; e += 256) {
    int j = e & (NP1 - 1);
    int c = e >> 5;
    xs[j * XS_STR + c] = (_Float16)x[((size_t)bb * C_IN + c) * NPIX + n0 + j];
  }
  __syncthreads();

  const unsigned* xu = (const unsigned*)xs;

  // ---- projection A (no softmax) ----
  {
    v8f a0, a1;
    proj2((const unsigned*)WAh, xu, m0, lm, g, a0, a1);
#pragma unroll
    for (int r = 0; r < 8; ++r) {
      int m = m0 + r + 8 * g;
      float bias = bA[m];
      Ag[((size_t)bb * CM + m) * NPIX + n0 + lm]      = (_Float16)(a0[r] + bias);
      Ag[((size_t)bb * CM + m) * NPIX + n0 + 16 + lm] = (_Float16)(a1[r] + bias);
    }
  }

  // ---- projection B + channel softmax ----
  {
    v8f a0, a1;
    proj2((const unsigned*)WBh, xu, m0, lm, g, a0, a1);
#pragma unroll
    for (int r = 0; r < 8; ++r) {
      int m = m0 + r + 8 * g;
      float bias = bB[m];
      bmS[lm * BM_STR + m]        = a0[r] + bias;
      bmS[(16 + lm) * BM_STR + m] = a1[r] + bias;
    }
    __syncthreads();
    softmax32x128_store(bmS, red, tid, Bmg, bb, n0);
    __syncthreads();
  }

  // ---- projection V + channel softmax ----
  {
    v8f a0, a1;
    proj2((const unsigned*)WVh, xu, m0, lm, g, a0, a1);
#pragma unroll
    for (int r = 0; r < 8; ++r) {
      int m = m0 + r + 8 * g;
      float bias = bV[m];
      bmS[lm * BM_STR + m]        = a0[r] + bias;
      bmS[(16 + lm) * BM_STR + m] = a1[r] + bias;
    }
    __syncthreads();
    softmax32x128_store(bmS, red, tid, Vmg, bb, n0);
  }
}

// ---------------------------------------------------------------------------
// Pass 1b: gd[b] = A[b] (128 x 4096) x Bm[b]^T (4096 x 128), f16 output.
// Grid: 16 batches x 8 m-tiles; wave w owns k-stripe [16w, 16w+16).
// Deterministic fixed-order reduction over n (no atomics).
// ---------------------------------------------------------------------------
__global__ __launch_bounds__(256) void k1b_gd(
    const _Float16* __restrict__ Ag, const _Float16* __restrict__ Bmg,
    _Float16* __restrict__ gdg)
{
  const int blk = blockIdx.x;
  const int bb  = blk >> 3;
  const int mt  = blk & 7;
  const int tid = threadIdx.x;
  const int w   = tid >> 5;
  const int lane = tid & 31;
  const int lm = lane & 15, g = lane >> 4;

  const unsigned* Au = (const unsigned*)Ag;
  const unsigned* Bu = (const unsigned*)Bmg;
  const size_t arow = ((size_t)bb * CM + mt * 16 + lm) * NPIX;
  const size_t brow = ((size_t)bb * CM + w * 16 + lm) * NPIX;

  v8f acc = {0.f,0.f,0.f,0.f,0.f,0.f,0.f,0.f};
#pragma unroll 4
  for (int nn = 0; nn < NPIX; nn += 32) {
    v16h a  = load_fragA(Au, arow + nn, g);
    v16h bf = load_fragB(Bu, brow + nn, g);
    acc = wmma_f16(a, bf, acc);
  }
#pragma unroll
  for (int r = 0; r < 8; ++r)
    gdg[((size_t)bb * CM + mt * 16 + r + 8 * g) * CM + w * 16 + lm] = (_Float16)acc[r];
}

// ---------------------------------------------------------------------------
// Pass 2: Z = gd x Vm, out = WR x Z + bR, fused via LDS.
// One workgroup = (batch, 64-pixel tile).
// ---------------------------------------------------------------------------
__global__ __launch_bounds__(256) void k2_out(
    const _Float16* __restrict__ Vmg, const _Float16* __restrict__ gdg,
    const _Float16* __restrict__ WRh, const float* __restrict__ bR,
    float* __restrict__ out)
{
  __shared__ __align__(16) _Float16 vmS[NP2 * VM_STR];  // 17,408 B
  __shared__ __align__(16) _Float16 zS[NP2 * VM_STR];   // 17,408 B

  const int blk = blockIdx.x;
  const int bb  = blk >> 6;                 // 64 tiles per batch
  const int n0  = (blk & 63) * NP2;
  const int tid = threadIdx.x;
  const int w   = tid >> 5;
  const int lane = tid & 31;
  const int lm = lane & 15, g = lane >> 4;
  const int m0 = w * 16;

  // Stage Vm tile transposed: vmS[n][k] (k contiguous for B-fragment loads)
  for (int e = tid; e < NP2 * CM; e += 256) {
    int k = e >> 6, j = e & 63;
    vmS[j * VM_STR + k] = Vmg[((size_t)bb * CM + k) * NPIX + n0 + j];
  }
  __syncthreads();

  const v8f zf = {0.f,0.f,0.f,0.f,0.f,0.f,0.f,0.f};

  // Z = gd (128x128) x Vm (128x64): wave = 16-row m-stripe, 4 n-subtiles
  {
    const unsigned* gu = (const unsigned*)gdg;
    const unsigned* vu = (const unsigned*)vmS;
    v8f accz[4] = {zf, zf, zf, zf};
    const size_t grow = ((size_t)bb * CM + m0 + lm) * CM;
#pragma unroll
    for (int k0 = 0; k0 < CM; k0 += 32) {
      v16h a = load_fragA(gu, grow + k0, g);
#pragma unroll
      for (int ns = 0; ns < 4; ++ns) {
        v16h bf = load_fragB(vu, (size_t)(16 * ns + lm) * VM_STR + k0, g);
        accz[ns] = wmma_f16(a, bf, accz[ns]);
      }
    }
#pragma unroll
    for (int ns = 0; ns < 4; ++ns)
#pragma unroll
      for (int r = 0; r < 8; ++r)
        zS[(16 * ns + lm) * VM_STR + m0 + r + 8 * g] = (_Float16)accz[ns][r];
  }
  __syncthreads();

  // out = WR (512x128) x Z (128x64) + bR : wave covers c in [64w, 64w+64)
  {
    const unsigned* wru = (const unsigned*)WRh;
    const unsigned* zu  = (const unsigned*)zS;
    for (int cs = 0; cs < 4; ++cs) {
      const int cbase = w * 64 + cs * 16;
      v8f acc[4] = {zf, zf, zf, zf};
#pragma unroll
      for (int k0 = 0; k0 < CM; k0 += 32) {
        v16h a = load_fragA(wru, (size_t)(cbase + lm) * CM + k0, g);
#pragma unroll
        for (int ns = 0; ns < 4; ++ns) {
          v16h bf = load_fragB(zu, (size_t)(16 * ns + lm) * VM_STR + k0, g);
          acc[ns] = wmma_f16(a, bf, acc[ns]);
        }
      }
#pragma unroll
      for (int ns = 0; ns < 4; ++ns)
#pragma unroll
        for (int r = 0; r < 8; ++r) {
          int c = cbase + r + 8 * g;
          out[((size_t)bb * 512 + c) * NPIX + n0 + 16 * ns + lm] = acc[ns][r] + bR[c];
        }
    }
  }
}

// ---------------------------------------------------------------------------
// Host launcher
// ---------------------------------------------------------------------------
extern "C" void kernel_launch(void* const* d_in, const int* in_sizes, int n_in,
                              void* d_out, int out_size, void* d_ws, size_t ws_size,
                              hipStream_t stream) {
  const float* x  = (const float*)d_in[0];
  const float* WA = (const float*)d_in[1];
  const float* bA = (const float*)d_in[2];
  const float* WB = (const float*)d_in[3];
  const float* bB = (const float*)d_in[4];
  const float* WV = (const float*)d_in[5];
  const float* bV = (const float*)d_in[6];
  const float* WR = (const float*)d_in[7];
  const float* bR = (const float*)d_in[8];

  char* ws = (char*)d_ws;
  const size_t W_HALVES = (size_t)CM * C_IN;           // 65,536 halves
  const size_t P_HALVES = (size_t)16 * CM * NPIX;      // 8,388,608 halves
  _Float16* WAh = (_Float16*)(ws);
  _Float16* WBh = WAh + W_HALVES;
  _Float16* WVh = WBh + W_HALVES;
  _Float16* WRh = WVh + W_HALVES;
  _Float16* Ag  = WRh + W_HALVES;
  _Float16* Bmg = Ag + P_HALVES;
  _Float16* Vmg = Bmg + P_HALVES;
  _Float16* gdg = Vmg + P_HALVES;                      // 16*128*128 halves

  (void)in_sizes; (void)n_in; (void)out_size; (void)ws_size;

  kprep<<<(CM * C_IN + 255) / 256, 256, 0, stream>>>(WA, WB, WV, WR,
                                                     WAh, WBh, WVh, WRh);
  k1a_proj<<<16 * (NPIX / NP1), 256, 0, stream>>>(x, bA, bB, bV,
                                                  WAh, WBh, WVh, Ag, Bmg, Vmg);
  k1b_gd<<<16 * 8, 256, 0, stream>>>(Ag, Bmg, gdg);
  k2_out<<<16 * (NPIX / NP2), 256, 0, stream>>>(Vmg, gdg, WRh, bR, (float*)d_out);
}
